// GATGNN_43069932044423
// MI455X (gfx1250) — matrix operand order repaired and checked by reading, not compiled, not run on hardware
//
#include <hip/hip_runtime.h>
#include <hip/hip_bf16.h>

#define IN_CH 512
#define HID   64
#define HEADS 4
#define NC    256   /* HEADS*HID == OUT_CH */
#define NEG_SLOPE 0.2f

typedef __attribute__((ext_vector_type(16))) _Float16 v16h;
typedef __attribute__((ext_vector_type(8)))  _Float16 v8h;
typedef __attribute__((ext_vector_type(8)))  float    v8f;

union frag16 { v16h v; v8h h[2]; };

// ---------------- helpers ----------------
__device__ __forceinline__ float lrelu(float v) { return v > 0.0f ? v : NEG_SLOPE * v; }

// order-preserving float->uint mapping for atomic max
__device__ __forceinline__ unsigned f2ord(float x) {
    unsigned u = __float_as_uint(x);
    return (u & 0x80000000u) ? ~u : (u | 0x80000000u);
}
__device__ __forceinline__ float ord2f(unsigned u) {
    return __uint_as_float((u & 0x80000000u) ? (u & 0x7FFFFFFFu) : ~u);
}

__device__ __forceinline__ void edge_sd(const int* ei, int E, int e, int& s, int& d) {
    if (e < E) { s = ei[e]; d = ei[E + e]; }
    else       { s = e - E; d = s; }           // self loops appended
}

// ---------------- elementwise / init ----------------
__global__ void k_f32_to_f16(const float* __restrict__ src, _Float16* __restrict__ dst, int n) {
    int i = blockIdx.x * blockDim.x + threadIdx.x;
    if (i < n) dst[i] = (_Float16)src[i];
}

// src[rows][cols] (f32, row-major) -> dst[cols][rows] (f16, i.e. column-major of src)
__global__ void k_f32_to_f16_T(const float* __restrict__ src, _Float16* __restrict__ dst,
                               int rows, int cols) {
    int i = blockIdx.x * blockDim.x + threadIdx.x;
    if (i >= rows * cols) return;
    int r = i / cols, c = i - r * cols;
    dst[(size_t)c * rows + r] = (_Float16)src[i];
}

__global__ void k_zero_f32(float* __restrict__ p, int n) {
    int i = blockIdx.x * blockDim.x + threadIdx.x;
    if (i < n) p[i] = 0.0f;
}

__global__ void k_init_node(unsigned* __restrict__ amax, float* __restrict__ denom,
                            float* __restrict__ deg, int n4, int n) {
    int i = blockIdx.x * blockDim.x + threadIdx.x;
    if (i < n4) { amax[i] = 0u; denom[i] = 0.0f; }
    if (i < n)  deg[i] = 0.0f;
}

__global__ void k_zero_pool(float* __restrict__ pool, float* __restrict__ cnt, int gnc, int g) {
    int i = blockIdx.x * blockDim.x + threadIdx.x;
    if (i < gnc) pool[i] = 0.0f;
    if (i < g)   cnt[i]  = 0.0f;
}

// ---------------- WMMA GEMM: C[M,Ncols] = A[M,K](f16) * BT[Ncols,K](f16, pre-transposed), f32 acc
// block = 256 threads = 8 waves; wave tile = 32 rows x 128 cols; K step = 32.
// Per K-step: 4 A loads + 16 B loads (all b128), then 16 back-to-back WMMAs
// (B fragments reused across both M sub-tiles).
__global__ __launch_bounds__(256)
void k_gemm_f16_wmma(const _Float16* __restrict__ A, const _Float16* __restrict__ BT,
                     float* __restrict__ C, int M, int K, int Ncols) {
    const int lane = threadIdx.x & 31;
    const int wave = threadIdx.x >> 5;
    const int laneLo   = lane & 15;
    const int laneHalf = lane >> 4;

    const int m0 = (blockIdx.x * 8 + wave) * 32;
    const int n0 = blockIdx.y * 128;
    if (m0 >= M) return;                     // wave-uniform: EXEC stays all-1s

    v8f acc0[8], acc1[8];
    #pragma unroll
    for (int t = 0; t < 8; ++t) {
        acc0[t] = (v8f){0,0,0,0,0,0,0,0};
        acc1[t] = (v8f){0,0,0,0,0,0,0,0};
    }

    int rowA1 = m0 + 16 + laneLo;
    if (rowA1 >= M) rowA1 = M - 1;           // clamp: loads stay in-bounds; stores guarded below
    const _Float16* Arow0 = A + (size_t)(m0 + laneLo) * K;
    const _Float16* Arow1 = A + (size_t)rowA1 * K;
    // B fragment base: lane holds column (n0 + t*16 + laneLo), K runs at k0 + laneHalf*8 and k0 + 16 + laneHalf*8
    const _Float16* Bcol = BT + (size_t)(n0 + laneLo) * K + laneHalf * 8;

    for (int k0 = 0; k0 < K; k0 += 32) {
        // A fragments 16x32: lanes 0-15 K [0..7 | 16..23]; lanes 16-31 K [8..15 | 24..31]
        frag16 a0, a1;
        a0.h[0] = *(const v8h*)(Arow0 + k0 + laneHalf * 8);
        a0.h[1] = *(const v8h*)(Arow0 + k0 + 16 + laneHalf * 8);
        a1.h[0] = *(const v8h*)(Arow1 + k0 + laneHalf * 8);
        a1.h[1] = *(const v8h*)(Arow1 + k0 + 16 + laneHalf * 8);

        // all 8 B fragments first (one wait point, then dense WMMA chain)
        frag16 b[8];
        #pragma unroll
        for (int t = 0; t < 8; ++t) {
            const _Float16* bp = Bcol + (size_t)t * 16 * K + k0;
            b[t].h[0] = *(const v8h*)(bp);
            b[t].h[1] = *(const v8h*)(bp + 16);
        }

        #pragma unroll
        for (int t = 0; t < 8; ++t)
            acc0[t] = __builtin_amdgcn_wmma_f32_16x16x32_f16(
                false, a0.v, false, b[t].v, (short)0, acc0[t], false, false);
        #pragma unroll
        for (int t = 0; t < 8; ++t)
            acc1[t] = __builtin_amdgcn_wmma_f32_16x16x32_f16(
                false, a1.v, false, b[t].v, (short)0, acc1[t], false, false);
    }

    // store: C/D layout — VGPR r: lanes 0-15 M=r, lanes 16-31 M=8+r; N=lane%16
    #pragma unroll
    for (int t = 0; t < 8; ++t) {
        const int col = n0 + t * 16 + laneLo;
        #pragma unroll
        for (int r = 0; r < 8; ++r)
            C[(size_t)(m0 + r + laneHalf * 8) * Ncols + col] = acc0[t][r];
    }
    if (m0 + 16 < M) {                       // wave-uniform guard for second M sub-tile
        #pragma unroll
        for (int t = 0; t < 8; ++t) {
            const int col = n0 + t * 16 + laneLo;
            #pragma unroll
            for (int r = 0; r < 8; ++r)
                C[(size_t)(m0 + 16 + r + laneHalf * 8) * Ncols + col] = acc1[t][r];
        }
    }
}

// ---------------- attention scores a_src / a_dst : [N, HEADS] ----------------
__global__ void k_attn_scores(const float* __restrict__ h1,
                              const float* __restrict__ att_src, const float* __restrict__ att_dst,
                              float* __restrict__ asrc, float* __restrict__ adst, int n) {
    int i = blockIdx.x * blockDim.x + threadIdx.x;
    if (i >= n * HEADS) return;
    int node = i >> 2, h = i & 3;
    const float* hp = h1 + (size_t)node * NC + h * HID;
    float s = 0.0f, d = 0.0f;
    #pragma unroll 8
    for (int c = 0; c < HID; ++c) {
        float v = hp[c];
        s += v * att_src[h * HID + c];
        d += v * att_dst[h * HID + c];
    }
    asrc[i] = s; adst[i] = d;
}

// ---------------- edge pass 1: segment max (ordered-uint atomicMax) + degree ----
__global__ void k_edge_amax_deg(const int* __restrict__ ei, int E, int n,
                                const float* __restrict__ asrc, const float* __restrict__ adst,
                                unsigned* __restrict__ amax, float* __restrict__ deg) {
    int e = blockIdx.x * blockDim.x + threadIdx.x;
    if (e >= E + n) return;
    int s, d; edge_sd(ei, E, e, s, d);
    atomicAdd(&deg[d], 1.0f);
    #pragma unroll
    for (int h = 0; h < HEADS; ++h) {
        float v = lrelu(asrc[s * HEADS + h] + adst[d * HEADS + h]);
        atomicMax(&amax[d * HEADS + h], f2ord(v));
    }
}

// ---------------- edge pass 2: denom = segment_sum(exp(alpha - amax)) ----------
__global__ void k_edge_denom(const int* __restrict__ ei, int E, int n,
                             const float* __restrict__ asrc, const float* __restrict__ adst,
                             const unsigned* __restrict__ amax, float* __restrict__ denom) {
    int e = blockIdx.x * blockDim.x + threadIdx.x;
    if (e >= E + n) return;
    int s, d; edge_sd(ei, E, e, s, d);
    #pragma unroll
    for (int h = 0; h < HEADS; ++h) {
        float v = lrelu(asrc[s * HEADS + h] + adst[d * HEADS + h]);
        float m = ord2f(amax[d * HEADS + h]);
        atomicAdd(&denom[d * HEADS + h], __expf(v - m));
    }
}

// ---------------- edge pass 3: out1[dst] += h1[src] * coef  (wave per edge) ----
__global__ void k_edge_message(const int* __restrict__ ei, int E, int n,
                               const float* __restrict__ asrc, const float* __restrict__ adst,
                               const unsigned* __restrict__ amax, const float* __restrict__ denom,
                               const float* __restrict__ h1, float* __restrict__ out1) {
    int gid = blockIdx.x * blockDim.x + threadIdx.x;
    int e = gid >> 5, lane = gid & 31;
    if (e >= E + n) return;
    int s, d; edge_sd(ei, E, e, s, d);
    int c0 = lane * 8;                 // 8 channels per lane, all in one head
    int h  = c0 >> 6;
    float v = lrelu(asrc[s * HEADS + h] + adst[d * HEADS + h]);
    float m = ord2f(amax[d * HEADS + h]);
    float coef = __expf(v - m) / (denom[d * HEADS + h] + 1e-16f);
    const float* hp = h1 + (size_t)s * NC + c0;
    float* op = out1 + (size_t)d * NC + c0;
    #pragma unroll
    for (int j = 0; j < 8; ++j) atomicAdd(&op[j], hp[j] * coef);
}

// ---------------- relu(out1 + b1) -> f16 for GEMM2 ----------------
__global__ void k_relu_bias_h16(const float* __restrict__ out1, const float* __restrict__ b1,
                                _Float16* __restrict__ h2h, int n) {
    int i = blockIdx.x * blockDim.x + threadIdx.x;
    if (i >= n * NC) return;
    float v = out1[i] + b1[i & (NC - 1)];
    h2h[i] = (_Float16)fmaxf(v, 0.0f);
}

__global__ void k_dinv(const float* __restrict__ deg, float* __restrict__ dinv, int n) {
    int i = blockIdx.x * blockDim.x + threadIdx.x;
    if (i < n) dinv[i] = rsqrtf(fmaxf(deg[i], 1.0f));
}

// ---------------- GCN aggregation: out2[dst] += h3[src] * dinv[s]*dinv[d] ------
__global__ void k_edge_gcn(const int* __restrict__ ei, int E, int n,
                           const float* __restrict__ dinv, const float* __restrict__ h3,
                           float* __restrict__ out2) {
    int gid = blockIdx.x * blockDim.x + threadIdx.x;
    int e = gid >> 5, lane = gid & 31;
    if (e >= E + n) return;
    int s, d; edge_sd(ei, E, e, s, d);
    float nrm = dinv[s] * dinv[d];
    int c0 = lane * 8;
    const float* hp = h3 + (size_t)s * NC + c0;
    float* op = out2 + (size_t)d * NC + c0;
    #pragma unroll
    for (int j = 0; j < 8; ++j) atomicAdd(&op[j], hp[j] * nrm);
}

// ---------------- mean pool: accumulate (out2 + b2) per graph ----------------
__global__ void k_pool_accum(const float* __restrict__ out2, const float* __restrict__ b2,
                             const int* __restrict__ batch, float* __restrict__ pool,
                             float* __restrict__ cnt, int n) {
    int gid = blockIdx.x * blockDim.x + threadIdx.x;
    int node = gid >> 5, lane = gid & 31;
    if (node >= n) return;
    int g = batch[node];
    int c0 = lane * 8;
    const float* xp = out2 + (size_t)node * NC + c0;
    float* pp = pool + (size_t)g * NC + c0;
    #pragma unroll
    for (int j = 0; j < 8; ++j) atomicAdd(&pp[j], xp[j] + b2[c0 + j]);
    if (lane == 0) atomicAdd(&cnt[g], 1.0f);
}

__global__ void k_finalize(const float* __restrict__ pool, const float* __restrict__ cnt,
                           float* __restrict__ out, int gnc) {
    int i = blockIdx.x * blockDim.x + threadIdx.x;
    if (i >= gnc) return;
    out[i] = pool[i] / fmaxf(cnt[i >> 8], 1.0f);
}

// =====================================================================
extern "C" void kernel_launch(void* const* d_in, const int* in_sizes, int n_in,
                              void* d_out, int out_size, void* d_ws, size_t ws_size,
                              hipStream_t stream) {
    const float* x       = (const float*)d_in[0];
    const int*   ei      = (const int*)  d_in[1];
    const int*   batch   = (const int*)  d_in[2];
    const float* W1      = (const float*)d_in[3];
    const float* att_src = (const float*)d_in[4];
    const float* att_dst = (const float*)d_in[5];
    const float* b1      = (const float*)d_in[6];
    const float* W2      = (const float*)d_in[7];
    const float* b2      = (const float*)d_in[8];
    float* out = (float*)d_out;

    const int N    = in_sizes[0] / IN_CH;
    const int E    = in_sizes[1] / 2;
    const int Etot = E + N;
    const int G    = out_size / NC;

    // ---- workspace carve-out (256B aligned) ----
    char* ws = (char*)d_ws;
    auto carve = [&](size_t bytes) -> char* {
        char* p = ws; ws += (bytes + 255) & ~(size_t)255; return p;
    };
    _Float16* xh   = (_Float16*)carve((size_t)N * IN_CH * 2);  // reused as out2 later
    float*    h1   = (float*)   carve((size_t)N * NC * 4);
    float*    out1 = (float*)   carve((size_t)N * NC * 4);     // reused as h3 later
    _Float16* h2h  = (_Float16*)carve((size_t)N * NC * 2);
    _Float16* W1T  = (_Float16*)carve((size_t)IN_CH * NC * 2); // [NC][IN_CH] col-major of W1
    _Float16* W2T  = (_Float16*)carve((size_t)NC * NC * 2);    // [NC][NC]   col-major of W2
    float*    asrc = (float*)   carve((size_t)N * HEADS * 4);
    float*    adst = (float*)   carve((size_t)N * HEADS * 4);
    unsigned* amax = (unsigned*)carve((size_t)N * HEADS * 4);
    float*    denm = (float*)   carve((size_t)N * HEADS * 4);
    float*    deg  = (float*)   carve((size_t)N * 4);
    float*    dinv = (float*)   carve((size_t)N * 4);
    float*    pool = (float*)   carve((size_t)G * NC * 4);
    float*    cnt  = (float*)   carve((size_t)G * 4);

    const int T = 256;
    auto blks = [&](long long n) { return (unsigned)((n + T - 1) / T); };

    // 1) fp32 -> fp16 conversions (weights transposed to column-major for vector B loads)
    k_f32_to_f16<<<blks((long long)N * IN_CH), T, 0, stream>>>(x, xh, N * IN_CH);
    k_f32_to_f16_T<<<blks(IN_CH * NC), T, 0, stream>>>(W1, W1T, IN_CH, NC);
    k_f32_to_f16_T<<<blks(NC * NC), T, 0, stream>>>(W2, W2T, NC, NC);

    // 2) GEMM1: h1 = x @ W1   (WMMA f16 -> f32)
    {
        dim3 grid((N + 255) / 256, NC / 128);
        k_gemm_f16_wmma<<<grid, 256, 0, stream>>>(xh, W1T, h1, N, IN_CH, NC);
    }

    // 3) attention scores
    k_attn_scores<<<blks((long long)N * HEADS), T, 0, stream>>>(h1, att_src, att_dst, asrc, adst, N);

    // 4) init per-node accumulators; edge passes
    k_init_node<<<blks((long long)N * HEADS), T, 0, stream>>>(amax, denm, deg, N * HEADS, N);
    k_edge_amax_deg<<<blks(Etot), T, 0, stream>>>(ei, E, N, asrc, adst, amax, deg);
    k_edge_denom<<<blks(Etot), T, 0, stream>>>(ei, E, N, asrc, adst, amax, denm);
    k_zero_f32<<<blks((long long)N * NC), T, 0, stream>>>(out1, N * NC);
    k_edge_message<<<blks((long long)Etot * 32), T, 0, stream>>>(ei, E, N, asrc, adst, amax, denm, h1, out1);

    // 5) relu + bias -> f16
    k_relu_bias_h16<<<blks((long long)N * NC), T, 0, stream>>>(out1, b1, h2h, N);

    // 6) degree norm
    k_dinv<<<blks(N), T, 0, stream>>>(deg, dinv, N);

    // 7) GEMM2: h3 = relu(h) @ W2   (reuse out1 buffer as h3)
    float* h3 = out1;
    {
        dim3 grid((N + 255) / 256, NC / 128);
        k_gemm_f16_wmma<<<grid, 256, 0, stream>>>(h2h, W2T, h3, N, NC, NC);
    }

    // 8) GCN aggregation (reuse xh region as out2)
    float* out2 = (float*)xh;
    k_zero_f32<<<blks((long long)N * NC), T, 0, stream>>>(out2, N * NC);
    k_edge_gcn<<<blks((long long)Etot * 32), T, 0, stream>>>(ei, E, N, dinv, h3, out2);

    // 9) mean pool
    k_zero_pool<<<blks((long long)G * NC), T, 0, stream>>>(pool, cnt, G * NC, G);
    k_pool_accum<<<blks((long long)N * 32), T, 0, stream>>>(out2, b2, batch, pool, cnt, N);
    k_finalize<<<blks((long long)G * NC), T, 0, stream>>>(pool, cnt, out, G * NC);
}
